// RaindropBackbone_3212635537848
// MI455X (gfx1250) — matrix operand order repaired and checked by reading, not compiled, not run on hardware
//
#include <hip/hip_runtime.h>
#include <hip/hip_bf16.h>
#include <math.h>

typedef __attribute__((ext_vector_type(16))) _Float16 v16h;
typedef __attribute__((ext_vector_type(8)))  float    v8f;

static constexpr int Bn = 32, Tn = 256, Dn = 64, Hn = 128, G3 = 384;

__device__ __forceinline__ int laneid() { return threadIdx.x & 31; }

// A-matrix fragment, 16x32 f16 (M x K), element (m,K) = base[m*ld + K].
// Lanes 0-15 -> rows, K 0..15; lanes 16-31 -> K 16..31 (contiguous, matches B).
__device__ __forceinline__ v16h load_a_frag(const _Float16* base, int ld) {
  const int l  = laneid();
  const int m  = l & 15;
  const int kb = (l < 16) ? 0 : 16;
  const _Float16* row = base + m * ld;
  v16h a;
#pragma unroll
  for (int i = 0; i < 16; ++i) a[i] = row[kb + i];
  return a;
}

// B-matrix fragment, 32x16 f16 (K x N), sourced from BT where BT[n][k] = B[k][n].
// Lane n = l&15; lanes 0-15 hold K=0..15, lanes 16-31 hold K=16..31 (contiguous in BT row).
__device__ __forceinline__ v16h load_b_frag(const _Float16* bt, int ld) {
  const int l  = laneid();
  const int n  = l & 15;
  const int kb = (l < 16) ? 0 : 16;
  const _Float16* row = bt + n * ld + kb;
  v16h b;
#pragma unroll
  for (int i = 0; i < 16; ++i) b[i] = row[i];
  return b;
}

__device__ __forceinline__ v8f wmma_f16(v16h a, v16h b, v8f c) {
  return __builtin_amdgcn_wmma_f32_16x16x32_f16(false, a, false, b, (short)0, c, false, false);
}

__device__ __forceinline__ float fsigmoid(float x) { return 1.f / (1.f + __expf(-x)); }
__device__ __forceinline__ float ftanh(float x)    { return 2.f / (1.f + __expf(-2.f * x)) - 1.f; }

// ---------------- prep kernels ----------------

__global__ void presence_kernel(const float* __restrict__ mmask, float* __restrict__ pres) {
  int id = blockIdx.x * 256 + threadIdx.x;
  if (id >= Bn * Dn) return;
  int b = id >> 6, d = id & 63;
  float s = 0.f;
  for (int t = 0; t < Tn; ++t) s += 1.f - mmask[((size_t)b * Tn + t) * Dn + d];
  pres[id] = s * (1.f / Tn);
}

__global__ void sim_kernel(const float* __restrict__ E, float* __restrict__ sim) {
  int id = blockIdx.x * 256 + threadIdx.x;
  if (id >= Dn * Dn) return;
  int i = id >> 6, j = id & 63;
  float s = 0.f;
  for (int h = 0; h < Hn; ++h) s += E[i * Hn + h] * E[j * Hn + h];
  sim[id] = fmaxf(s, 0.f);
}

// Fold row-normalized adjacency + mean over i into w[b,j].
__global__ void __launch_bounds__(64) w_kernel(const float* __restrict__ pres,
                                               const float* __restrict__ sim,
                                               float* __restrict__ w) {
  __shared__ float sp[64], srs[64];
  const int b = blockIdx.x, i = threadIdx.x;
  sp[i] = pres[b * 64 + i];
  __syncthreads();
  float s = 0.f;
  for (int j = 0; j < 64; ++j) s += sp[j] * sim[i * 64 + j];
  float rs = fmaxf(sp[i] * s + 1.f, 1e-6f);   // row-sum of adj row i (incl. identity)
  srs[i] = rs;
  __syncthreads();
  float acc = 0.f;                            // thread index now plays column j
  for (int k = 0; k < 64; ++k) acc += sp[k] * sim[k * 64 + i] / srs[k];
  w[b * 64 + i] = (sp[i] * acc + 1.f / srs[i]) * (1.f / 64.f);
}

__global__ void et_kernel(const float* __restrict__ E, _Float16* __restrict__ et) {
  int id = blockIdx.x * 256 + threadIdx.x;             // ET[n][k] = E[k][n], [128][64] f16
  if (id >= Hn * Dn) return;
  int n = id >> 6, k = id & 63;
  et[id] = (_Float16)E[k * Hn + n];
}

__global__ void cvt_kernel(const float* __restrict__ wih, const float* __restrict__ whh,
                           _Float16* __restrict__ wih16, _Float16* __restrict__ whh16) {
  int id = blockIdx.x * 256 + threadIdx.x;
  const int N = G3 * Hn;
  if (id < N)          wih16[id]     = (_Float16)wih[id];
  else if (id < 2 * N) whh16[id - N] = (_Float16)whh[id - N];
}

// ---------------- GEMM 1: visit_repr = (w*x*obs) @ E + time_enc, f16 out ----------------

__global__ void __launch_bounds__(256) gemm1_kernel(const float* __restrict__ x,
                                                    const float* __restrict__ mmask,
                                                    const float* __restrict__ vt,
                                                    const float* __restrict__ wj,
                                                    const _Float16* __restrict__ et,
                                                    _Float16* __restrict__ vr) {
  __shared__ _Float16 s_c[16 * 64];
  const int tid = threadIdx.x;
  const int blk = blockIdx.x;                 // 16-row strip of M=8192
#pragma unroll
  for (int k = 0; k < 4; ++k) {
    int e = tid + 256 * k;
    int m = e >> 6, j = e & 63;
    int g = blk * 16 + m;
    int b = g >> 8;
    size_t off = (size_t)g * 64 + j;
    s_c[m * 64 + j] = (_Float16)(wj[b * 64 + j] * x[off] * (1.f - mmask[off]));
  }
  __syncthreads();
  const int wave = tid >> 5;                  // n_tile, 8 waves cover N=128
  v16h a0 = load_a_frag(s_c + 0, 64);
  v16h a1 = load_a_frag(s_c + 32, 64);
  v8f acc = {};
  acc = wmma_f16(a0, load_b_frag(et + wave * 16 * 64 + 0, 64), acc);
  acc = wmma_f16(a1, load_b_frag(et + wave * 16 * 64 + 32, 64), acc);

  const int l = tid & 31;
  const int lm = l & 15, hi = (l >= 16) ? 8 : 0;
  const int col = wave * 16 + lm;
  const bool is_sin = (col < 64);                      // wave-uniform (tiles don't straddle 64)
  const float kf = (float)(is_sin ? col : col - 64);
  const float freq = __expf(-kf * 0.14391157f);        // ln(10000)/64
#pragma unroll
  for (int r = 0; r < 8; ++r) {
    int g = blk * 16 + r + hi;
    float ang = vt[g] * freq;
    float sv = __sinf(ang), cv = __cosf(ang);          // branchless: v_cndmask, no exec juggling
    float te = is_sin ? sv : cv;
    vr[(size_t)g * Hn + col] = (_Float16)(acc[r] + te);
  }
}

// ---------------- GEMM 2: gi[t][b][384] = visit_repr @ W_ih^T + b_ih ----------------

__global__ void __launch_bounds__(256) gemm2_kernel(const _Float16* __restrict__ vr,
                                                    const _Float16* __restrict__ wih16,
                                                    const float* __restrict__ bih,
                                                    float* __restrict__ gi) {
  const int tid = threadIdx.x;
  const int wave = tid >> 5;
  const int n_tile = blockIdx.y * 8 + wave;   // 0..23
  const int row0 = blockIdx.x * 16;
  v8f acc = {};
#pragma unroll
  for (int kc = 0; kc < 4; ++kc) {
    v16h a = load_a_frag(vr + (size_t)row0 * Hn + kc * 32, Hn);
    v16h b = load_b_frag(wih16 + (size_t)n_tile * 16 * Hn + kc * 32, Hn);
    acc = wmma_f16(a, b, acc);
  }
  const int l = tid & 31;
  const int lm = l & 15, hi = (l >= 16) ? 8 : 0;
  const int col = n_tile * 16 + lm;
  const float bias = bih[col];
#pragma unroll
  for (int r = 0; r < 8; ++r) {
    int g = row0 + r + hi;
    int b = g >> 8, t = g & 255;
    gi[((size_t)t * Bn + b) * G3 + col] = acc[r] + bias;   // time-major for recurrence
  }
}

// ---------------- persistent GRU recurrence: single workgroup, LDS-resident ----------------

__global__ void __launch_bounds__(256) gru_kernel(const float* __restrict__ gi,
                                                  const _Float16* __restrict__ whh16,
                                                  const float* __restrict__ bhh,
                                                  const int* __restrict__ lens,
                                                  float* __restrict__ out) {
  extern __shared__ char smem[];
  _Float16* s_whh = (_Float16*)(smem);             // [384][128] f16  (96 KB)
  _Float16* s_h   = (_Float16*)(smem + 98304);     // [32][128]  f16  (8 KB)
  float*    s_gh  = (float*)   (smem + 106496);    // [32][384]  f32  (48 KB)
  float*    s_bhh = (float*)   (smem + 155648);    // [384]
  int*      s_len = (int*)     (smem + 157184);    // [32]
  const int tid = threadIdx.x;

  { // stage W_hh into LDS as 16-byte chunks
    const uint4* src = (const uint4*)whh16;
    uint4* dst = (uint4*)s_whh;
#pragma unroll
    for (int k = 0; k < 24; ++k) dst[tid + 256 * k] = src[tid + 256 * k];
  }
  for (int k = tid; k < Bn * Hn; k += 256) s_h[k] = (_Float16)0.f;
  for (int k = tid; k < G3; k += 256) s_bhh[k] = bhh[k];
  if (tid < Bn) s_len[tid] = lens[tid];
  __syncthreads();

  const int wave = tid >> 5;
  const int l = tid & 31;
  const int m_tile = wave & 1;                 // batch rows 0-15 / 16-31
  const int n_base = wave >> 1;                // 4 wave-pairs x 6 n-tiles = 24
  const int lm = l & 15;
  const int hi = (l >= 16) ? 8 : 0;

  for (int t = 0; t < Tn; ++t) {
    const float* git = gi + (size_t)t * (Bn * G3);
    if (t + 1 < Tn) {                          // prefetch next step's 48KB gi slab,
      const float* nxt = git + Bn * G3;        // exact 128B-line coverage (384 lines)
      __builtin_prefetch(nxt + tid * 32, 0, 1);
      if (tid < 128) __builtin_prefetch(nxt + (256 + tid) * 32, 0, 1);
    }

    // gh = h @ W_hh^T  (K=128 as 4 chained K32 WMMAs)
    v16h a0 = load_a_frag(s_h + m_tile * 16 * Hn + 0,  Hn);
    v16h a1 = load_a_frag(s_h + m_tile * 16 * Hn + 32, Hn);
    v16h a2 = load_a_frag(s_h + m_tile * 16 * Hn + 64, Hn);
    v16h a3 = load_a_frag(s_h + m_tile * 16 * Hn + 96, Hn);
#pragma unroll
    for (int j = 0; j < 6; ++j) {
      const int n_tile = n_base + 4 * j;
      const _Float16* bt = s_whh + n_tile * 16 * Hn;
      v8f acc = {};
      acc = wmma_f16(a0, load_b_frag(bt + 0,  Hn), acc);
      acc = wmma_f16(a1, load_b_frag(bt + 32, Hn), acc);
      acc = wmma_f16(a2, load_b_frag(bt + 64, Hn), acc);
      acc = wmma_f16(a3, load_b_frag(bt + 96, Hn), acc);
      const int nn = n_tile * 16 + lm;
#pragma unroll
      for (int r = 0; r < 8; ++r) {
        const int mrow = m_tile * 16 + r + hi;
        s_gh[mrow * G3 + nn] = acc[r];
      }
    }
    __syncthreads();

    // fused gates + state update + masked output store
#pragma unroll
    for (int k = 0; k < 16; ++k) {
      const int idx = tid + 256 * k;           // 32*128 elements, 1:1 per thread
      const int b = idx >> 7, hc = idx & 127;
      const float ghr = s_gh[b * G3 + hc]        + s_bhh[hc];
      const float ghz = s_gh[b * G3 + 128 + hc]  + s_bhh[128 + hc];
      const float ghn = s_gh[b * G3 + 256 + hc]  + s_bhh[256 + hc];
      const float r = fsigmoid(git[b * G3 + hc]       + ghr);
      const float z = fsigmoid(git[b * G3 + 128 + hc] + ghz);
      const float n = ftanh(git[b * G3 + 256 + hc] + r * ghn);
      const float hold = (float)s_h[b * Hn + hc];
      const float hnew = (1.f - z) * n + z * hold;
      s_h[b * Hn + hc] = (_Float16)hnew;
      out[((size_t)b * Tn + t) * Hn + hc] = (t < s_len[b]) ? hnew : 0.f;
    }
    __syncthreads();
  }
}

// ---------------- launcher ----------------

extern "C" void kernel_launch(void* const* d_in, const int* in_sizes, int n_in,
                              void* d_out, int out_size, void* d_ws, size_t ws_size,
                              hipStream_t stream) {
  (void)in_sizes; (void)n_in; (void)out_size; (void)ws_size;
  const float* x     = (const float*)d_in[0];
  const float* mmask = (const float*)d_in[1];
  const float* vt    = (const float*)d_in[2];
  /* d_in[3] time_delta: unused by the reference */
  const int*   lens  = (const int*)d_in[4];
  const float* E     = (const float*)d_in[5];
  const float* wih   = (const float*)d_in[6];
  const float* whh   = (const float*)d_in[7];
  const float* bih   = (const float*)d_in[8];
  const float* bhh   = (const float*)d_in[9];
  float* out = (float*)d_out;

  char* ws = (char*)d_ws;
  float*    pres  = (float*)   (ws + 0);        // 8 KB
  float*    sim   = (float*)   (ws + 8192);     // 16 KB
  float*    wj    = (float*)   (ws + 24576);    // 8 KB
  _Float16* et    = (_Float16*)(ws + 32768);    // E^T f16, 16 KB
  _Float16* wih16 = (_Float16*)(ws + 49152);    // 96 KB
  _Float16* whh16 = (_Float16*)(ws + 147456);   // 96 KB
  _Float16* vr    = (_Float16*)(ws + 245760);   // visit_repr f16, 2 MB
  float*    gi    = (float*)   (ws + 2342912);  // input gates f32, 12 MB

  presence_kernel<<<8, 256, 0, stream>>>(mmask, pres);
  sim_kernel<<<16, 256, 0, stream>>>(E, sim);
  w_kernel<<<32, 64, 0, stream>>>(pres, sim, wj);
  et_kernel<<<32, 256, 0, stream>>>(E, et);
  cvt_kernel<<<384, 256, 0, stream>>>(wih, whh, wih16, whh16);
  gemm1_kernel<<<512, 256, 0, stream>>>(x, mmask, vt, wj, et, vr);
  gemm2_kernel<<<dim3(512, 3, 1), 256, 0, stream>>>(vr, wih16, bih, gi);
  gru_kernel<<<1, 256, 157312, stream>>>(gi, whh16, bhh, lens, out);
}
